// Interaction_16844861735177
// MI455X (gfx1250) — matrix operand order repaired and checked by reading, not compiled, not run on hardware
//
#include <hip/hip_runtime.h>
#include <hip/hip_fp16.h>

// ---------------------------------------------------------------------------
// MI455X (gfx1250) fused interaction block, round 3.
// ~58 GFLOP at ~1000 FLOP/byte -> compute bound -> v_wmma_f32_16x16x32_f16.
// This round: SiLU via v_rcp_f32 (was a 10-op IEEE divide chain in the
// cfconv hot loop and all GEMM epilogues).
// ---------------------------------------------------------------------------

typedef __attribute__((ext_vector_type(16))) _Float16 v16h;
typedef __attribute__((ext_vector_type(8)))  float    v8f;
typedef __attribute__((ext_vector_type(4)))  float    v4f;

constexpr int NA = 512;   // atoms
constexpr int NF = 128;   // features
constexpr int NK = 50;    // rbf
constexpr int KP = 64;    // NK padded
constexpr int NN = 511;   // neighbors per atom
constexpr int NH = 4;     // heads
constexpr int HD = 32;    // head dim
constexpr int C2 = 256;   // cat width
constexpr int NO = 384;   // ocf / o / u output width

// fast SiLU: x * v_rcp(1 + e^-x)   (v_exp_f32 + v_rcp_f32, no IEEE divide)
__device__ __forceinline__ float silu_f(float x) {
  return x * __builtin_amdgcn_rcpf(1.0f + __expf(-x));
}

union AFrag  { v16h v; v4f f4[2]; };
union Pack4h { _Float16 h[4]; unsigned long long u64; };

// A (16x32 f16) fragment from an LDS tile; row stride in halves (mult of 8).
__device__ __forceinline__ v16h load_a_frag_lds(const _Float16* tile, int strideH, int lane) {
  const int row = lane & 15;
  const int hf  = lane >> 4;
  const _Float16* base = tile + row * strideH;
  AFrag u;
  u.f4[0] = *(const v4f*)(base + 8 * hf);        // K = 8*hf .. +7
  u.f4[1] = *(const v4f*)(base + 16 + 8 * hf);   // K = 16+8*hf .. +7
  return u.v;
}

__device__ __forceinline__ v8f wmma16(v16h a, v16h b, v8f c) {
  return __builtin_amdgcn_wmma_f32_16x16x32_f16(false, a, false, b,
                                                (short)0, c, false, false);
}

// ---------------------------------------------------------------------------
// fp32 -> f16 bulk convert (4 elements / thread, packed 8B stores)
// ---------------------------------------------------------------------------
__global__ void to_half_kernel(const float* __restrict__ src,
                               _Float16* __restrict__ dst, int n4) {
  int idx = blockIdx.x * 256 + threadIdx.x;
  if (idx >= n4) return;
  const float* p = src + idx * 4;
  v4f f = *(const v4f*)p;
  Pack4h pk;
  pk.h[0] = (_Float16)f.x; pk.h[1] = (_Float16)f.y;
  pk.h[2] = (_Float16)f.z; pk.h[3] = (_Float16)f.w;
  *(unsigned long long*)(dst + idx * 4) = pk.u64;
}

// ---------------------------------------------------------------------------
// Transpose fp32 weight W[O][C] -> f16 T[Kp][O], zero-padded K rows.
// ---------------------------------------------------------------------------
__global__ void transpose_h_kernel(const float* __restrict__ W,
                                   _Float16* __restrict__ T,
                                   int O, int C, int Kp) {
  int idx = blockIdx.x * 256 + threadIdx.x;
  if (idx >= Kp * O) return;
  int kk = idx / O, o = idx - kk * O;
  float v = (kk < C) ? W[o * C + kk] : 0.0f;
  T[kk * O + o] = (_Float16)v;
}

// ---------------------------------------------------------------------------
// Generic WMMA GEMM, K fixed = 128, A already f16 row-major.
// No LDS, no barriers: each lane streams its own A/B fragments (b128 loads).
// C = act(A @ Bt + bias) (+add0) (+add1); f32 and/or f16 outputs.
// block: 256 thr (8 waves) -> tile 16(M) x 128(N).
// ---------------------------------------------------------------------------
__global__ void __launch_bounds__(256)
gemm_h_kernel(const _Float16* __restrict__ A, const _Float16* __restrict__ Bt,
              const float* __restrict__ bias, const float* __restrict__ add0,
              const float* __restrict__ add1, float* __restrict__ C,
              _Float16* __restrict__ Ch, int N, int act) {
  const int tid  = threadIdx.x;
  const int wave = tid >> 5, lane = tid & 31;
  const int m0   = blockIdx.x * 16;
  const int n0   = blockIdx.y * 128 + wave * 16;
  const int row  = lane & 15, hf = lane >> 4;

  const _Float16* arow = A + (m0 + row) * NF + 8 * hf;   // lane's A row slice
  const _Float16* brow = Bt + n0 + row;                  // lane's B column (n=row)

  v8f acc = {};
#pragma unroll
  for (int ks = 0; ks < 4; ++ks) {                       // K = 128 = 4 x 32
    AFrag ua;
    ua.f4[0] = *(const v4f*)(arow + ks * 32);
    ua.f4[1] = *(const v4f*)(arow + ks * 32 + 16);
    v16h b;
    for (int s = 0; s < 16; ++s) b[s] = brow[(ks * 32 + (s & 7) + 8 * hf + 16 * (s >> 3)) * N];
    acc = wmma16(ua.v, b, acc);
  }

  const int n = n0 + row;
  float bv = bias ? bias[n] : 0.0f;
#pragma unroll
  for (int r = 0; r < 8; ++r) {
    int m = m0 + hf * 8 + r;
    float x = acc[r] + bv;
    if (act == 1) x = silu_f(x);
    int off = m * N + n;
    if (add0) x += add0[off];
    if (add1) x += add1[off];
    if (C)  C[off]  = x;
    if (Ch) Ch[off] = (_Float16)x;
  }
}

// ---------------------------------------------------------------------------
// Fused CFConv: one block per atom i, 8 waves, loop over 16-neighbor tiles.
// e->w1,w2 (WMMA) -> silu/gather/mask -> cat (LDS f16) -> cat@ocf (WMMA)
// -> per-lane register reductions (lane owns a fixed output column).
// ---------------------------------------------------------------------------
__global__ void __launch_bounds__(256)
cfconv_kernel(const float* __restrict__ e_g, const float* __restrict__ vn_g,
              const float* __restrict__ mask_g, const float* __restrict__ xp,
              const _Float16* __restrict__ w1T, const _Float16* __restrict__ w2T,
              const _Float16* __restrict__ ocfT,
              const float* __restrict__ w1_b, const float* __restrict__ w2_b,
              const float* __restrict__ ocf_b,
              float* __restrict__ s1o, float* __restrict__ s2o,
              float* __restrict__ vm3o) {
  const int i    = blockIdx.x;
  const int tid  = threadIdx.x;
  const int wave = tid >> 5, lane = tid & 31;
  const int hf   = lane >> 4, nn = lane & 15;

  // hoisted 64-bit bases; all inner indexing is 32-bit
  const float* eBase  = e_g    + (size_t)i * NN * NK;
  const float* vnBase = vn_g   + (size_t)i * NN * 3;
  const float* mBase  = mask_g + (size_t)i * NN;

  __shared__ _Float16 A1[16 * 72];    // e tile (16 x 64, padded stride)
  __shared__ _Float16 A2[16 * 264];   // cat tile (16 x 256, padded stride)
  __shared__ float    xpi[NF];
  __shared__ float    mrow[16];
  __shared__ float    accO[C2 + 3 * NF];   // s1 | s2 | vm3[3][128]

  if (tid < NF) xpi[tid] = xp[i * NF + tid];
  for (int t = tid; t < C2 + 3 * NF; t += 256) accO[t] = 0.0f;

  const int n1 = wave * 16;           // GEMM1 column tile (w1 & w2)
  float s12p[3] = {0.0f, 0.0f, 0.0f};
  float vmp[3][3] = {};

  // staging coords: one thread = 4 consecutive rbf of one row
  const int sr  = tid >> 4;           // row 0..15
  const int sc4 = (tid & 15) * 4;     // col 0,4,...,60

  for (int jt = 0; jt < 32; ++jt) {
    const int j0 = jt * 16;
    __syncthreads();                  // protect A1/A2/mrow reuse
    if (tid < 16) {
      int jg = j0 + tid;
      mrow[tid] = (jg < NN) ? mBase[jg] : 0.0f;
    }
    {
      int jg = j0 + sr;
      float f0 = 0.f, f1 = 0.f, f2 = 0.f, f3 = 0.f;
      if (jg < NN) {
        const float* p = eBase + jg * NK + sc4;
        if (sc4 + 4 <= NK) {          // full quad (sc4 <= 44)
          float2 a0 = *(const float2*)p;
          float2 a1 = *(const float2*)(p + 2);
          f0 = a0.x; f1 = a0.y; f2 = a1.x; f3 = a1.y;
        } else if (sc4 < NK) {        // fringe quad at 48 (48,49 valid)
          f0 = p[0]; f1 = p[1];
        }
      }
      Pack4h pk;
      pk.h[0] = (_Float16)f0; pk.h[1] = (_Float16)f1;
      pk.h[2] = (_Float16)f2; pk.h[3] = (_Float16)f3;
      *(unsigned long long*)&A1[sr * 72 + sc4] = pk.u64;
    }
    if (jt + 1 < 32 && tid < 16) {    // L2 prefetch hint for next e tile
      int jg2 = j0 + 16 + tid;
      if (jg2 < NN) __builtin_prefetch(eBase + jg2 * NK, 0, 1);
    }
    __syncthreads();

    // ---- GEMM1: e(16x64) @ {w1T,w2T}(64x128) ----
    v8f d1 = {}, d2 = {};
#pragma unroll
    for (int ks = 0; ks < 2; ++ks) {
      v16h a  = load_a_frag_lds(A1 + 32 * ks, 72, lane);
      v16h b1;
      for (int s = 0; s < 16; ++s) b1[s] = w1T[(32 * ks + (s & 7) + 8 * hf + 16 * (s >> 3)) * NF + n1 + nn];
      d1 = wmma16(a, b1, d1);
      v16h b2;
      for (int s = 0; s < 16; ++s) b2[s] = w2T[(32 * ks + (s & 7) + 8 * hf + 16 * (s >> 3)) * NF + n1 + nn];
      d2 = wmma16(a, b2, d2);
    }
    // ---- epilogue: silu, xp / xp[nbr] gather, mask -> cat tile in LDS ----
    {
      float b1v = w1_b[n1 + nn], b2v = w2_b[n1 + nn];
      float xv  = xpi[n1 + nn];
#pragma unroll
      for (int r = 0; r < 8; ++r) {
        int m  = hf * 8 + r;
        int jg = j0 + m;
        int nb = (jg < i) ? jg : jg + 1;
        if (jg >= NN) nb = 0;
        float mk  = mrow[m];
        float c1  = silu_f(d1[r] + b1v) * xv * mk;
        float xnb = xp[nb * NF + n1 + nn];
        float c2  = silu_f(d2[r] + b2v) * xnb * mk;
        A2[m * 264 + n1 + nn]      = (_Float16)c1;
        A2[m * 264 + NF + n1 + nn] = (_Float16)c2;
      }
    }
    __syncthreads();

    // ---- GEMM2: cat(16x256) @ ocfT(256x384), 3 n-tiles per wave ----
    v8f acc2[3];
#pragma unroll
    for (int t = 0; t < 3; ++t) acc2[t] = v8f{};
#pragma unroll
    for (int ks = 0; ks < 8; ++ks) {
      v16h a = load_a_frag_lds(A2 + 32 * ks, 264, lane);
#pragma unroll
      for (int t = 0; t < 3; ++t) {
        int n0 = wave * 48 + t * 16;
        v16h b;
        for (int s = 0; s < 16; ++s) b[s] = ocfT[(32 * ks + (s & 7) + 8 * hf + 16 * (s >> 3)) * NO + n0 + nn];
        acc2[t] = wmma16(a, b, acc2[t]);
      }
    }
    // ---- per-lane register reductions over neighbors ----
#pragma unroll
    for (int t = 0; t < 3; ++t) {
      int n_abs = wave * 48 + t * 16 + nn;  // tile is region-uniform
      float bo = ocf_b[n_abs];
      if (n_abs < C2) {                     // s1 / s2: plain sum over j
        float p = 0.0f;
#pragma unroll
        for (int r = 0; r < 8; ++r) {
          int m = hf * 8 + r;
          p += (acc2[t][r] + bo) * mrow[m];
        }
        s12p[t] += p;
      } else {                              // s3: contract with vec_norm
#pragma unroll
        for (int r = 0; r < 8; ++r) {
          int m  = hf * 8 + r;
          int jg = j0 + m;
          float val = (acc2[t][r] + bo) * mrow[m];   // 0 when jg>=NN
          int vb = ((jg < NN) ? jg : 0) * 3;
#pragma unroll
          for (int d = 0; d < 3; ++d)
            vmp[t][d] += val * vnBase[vb + d];
        }
      }
    }
  }
  __syncthreads();
  // one LDS atomic per lane per owned column (2 lanes share a column)
#pragma unroll
  for (int t = 0; t < 3; ++t) {
    int n_abs = wave * 48 + t * 16 + nn;
    if (n_abs < C2) {
      atomicAdd(&accO[n_abs], s12p[t]);
    } else {
      int f = n_abs - C2;
#pragma unroll
      for (int d = 0; d < 3; ++d)
        atomicAdd(&accO[C2 + d * NF + f], vmp[t][d]);
    }
  }
  __syncthreads();
  if (tid < NF) {
    s1o[i * NF + tid] = accO[tid];
    s2o[i * NF + tid] = accO[NF + tid];
  }
  for (int t = tid; t < 3 * NF; t += 256)
    vm3o[i * 3 * NF + t] = accO[C2 + t];
}

// ---------------------------------------------------------------------------
// Attention: one block per query atom, one wave32 per head, two-pass softmax (max first, then exp-sum).
// ---------------------------------------------------------------------------
__global__ void __launch_bounds__(128)
attn_kernel(const float* __restrict__ q, const float* __restrict__ k,
            const float* __restrict__ va, const unsigned char* __restrict__ bmask,
            float* __restrict__ s_nl) {
  const int i = blockIdx.x;
  const int h = threadIdx.x >> 5, lane = threadIdx.x & 31;
  __shared__ float qsh[NH * HD];
  __shared__ float oacc[NH * HD];
  __shared__ float ssum[NH];
  qsh[threadIdx.x]  = q[i * NF + threadIdx.x];
  oacc[threadIdx.x] = 0.0f;
  __syncthreads();
  const float* qh = &qsh[h * HD];
  const unsigned char* bm = bmask + i * NA;
  float aloc[16];
  float mx = -3.0e38f;
#pragma unroll
  for (int jj = 0; jj < 16; ++jj) {
    int j = lane + jj * 32;
    const float* kr = k + j * NF + h * HD;
    float dot = 0.0f;
#pragma unroll
    for (int d = 0; d < HD; ++d) dot += qh[d] * kr[d];
    dot *= 0.125f;                          // 1/TEMP, TEMP = sqrt(64) = 8
    if (bm[j]) dot = -3.0e38f;
    aloc[jj] = dot;
    mx = fmaxf(mx, dot);
  }
#pragma unroll
  for (int s = 16; s > 0; s >>= 1) mx = fmaxf(mx, __shfl_xor(mx, s, 32));
  float se = 0.0f;
  float vacc[HD];
#pragma unroll
  for (int d = 0; d < HD; ++d) vacc[d] = 0.0f;
#pragma unroll
  for (int jj = 0; jj < 16; ++jj) {
    int j = lane + jj * 32;
    float p = __expf(aloc[jj] - mx);
    se += p;
    const float* vr = va + j * NF + h * HD;
#pragma unroll
    for (int d = 0; d < HD; ++d) vacc[d] += p * vr[d];
  }
#pragma unroll
  for (int s = 16; s > 0; s >>= 1) se += __shfl_xor(se, s, 32);
  if (lane == 0) ssum[h] = se;
#pragma unroll
  for (int d = 0; d < HD; ++d) atomicAdd(&oacc[h * HD + d], vacc[d]);
  __syncthreads();
  s_nl[i * NF + threadIdx.x] = oacc[threadIdx.x] * __builtin_amdgcn_rcpf(ssum[h]);
}

// ---------------------------------------------------------------------------
// s_pre = s + s_nl + s1sum, written directly as f16 (feeds o_W GEMM only)
// ---------------------------------------------------------------------------
__global__ void spre_kernel(const float* __restrict__ s,
                            const float* __restrict__ s_nl,
                            const float* __restrict__ s1,
                            _Float16* __restrict__ spre16) {
  int idx = blockIdx.x * 256 + threadIdx.x;
  spre16[idx] = (_Float16)(s[idx] + s_nl[idx] + s1[idx]);
}

// ---------------------------------------------------------------------------
// Combine: s_m (+f16 copy for the residual MLP) and vector output v_m.
// ---------------------------------------------------------------------------
__global__ void __launch_bounds__(128)
combine_kernel(const float* __restrict__ s_n, const float* __restrict__ u_out,
               const float* __restrict__ s2sum, const float* __restrict__ vm3,
               const float* __restrict__ v, float* __restrict__ out_sm,
               _Float16* __restrict__ sm16, float* __restrict__ out_vm) {
  int i = blockIdx.x, f = threadIdx.x;
  const float* sni = s_n + i * NO;
  float sn1 = sni[f], sn2 = sni[NF + f], sn3 = sni[2 * NF + f];
  float dotv = 0.0f;
#pragma unroll
  for (int d = 0; d < 3; ++d) {
    const float* ur = u_out + (i * 3 + d) * NO;
    dotv += ur[f] * ur[NF + f];
  }
  float sm = sn1 + sn2 * dotv;
  out_sm[i * NF + f] = sm;
  sm16[i * NF + f]   = (_Float16)sm;
  float s2v = s2sum[i * NF + f];
#pragma unroll
  for (int d = 0; d < 3; ++d) {
    const float* ur = u_out + (i * 3 + d) * NO;
    int off = (i * 3 + d) * NF + f;
    out_vm[off] = sn3 * ur[2 * NF + f] + s2v * v[off] + vm3[off];
  }
}

// ---------------------------------------------------------------------------
// Launch
// ---------------------------------------------------------------------------
extern "C" void kernel_launch(void* const* d_in, const int* in_sizes, int n_in,
                              void* d_out, int out_size, void* d_ws, size_t ws_size,
                              hipStream_t stream) {
  const float* s        = (const float*)d_in[0];
  const float* o_prev   = (const float*)d_in[1];
  const float* v        = (const float*)d_in[2];
  const float* e        = (const float*)d_in[3];
  const float* vec_norm = (const float*)d_in[4];
  const float* mask     = (const float*)d_in[5];
  const float* w1_W = (const float*)d_in[6];  const float* w1_b = (const float*)d_in[7];
  const float* w2_W = (const float*)d_in[8];  const float* w2_b = (const float*)d_in[9];
  const float* phi_W = (const float*)d_in[10]; const float* phi_b = (const float*)d_in[11];
  const float* ocf_W = (const float*)d_in[12]; const float* ocf_b = (const float*)d_in[13];
  const float* q_W = (const float*)d_in[14];  const float* q_b = (const float*)d_in[15];
  const float* k_W = (const float*)d_in[16];  const float* k_b = (const float*)d_in[17];
  const float* vl_W = (const float*)d_in[18]; const float* vl_b = (const float*)d_in[19];
  const float* u_W = (const float*)d_in[20];
  const float* o_W = (const float*)d_in[21];  const float* o_b = (const float*)d_in[22];
  const float* r1_W = (const float*)d_in[23]; const float* r1_b = (const float*)d_in[24];
  const float* r2_W = (const float*)d_in[25]; const float* r2_b = (const float*)d_in[26];
  const unsigned char* bmask = (const unsigned char*)d_in[28];  // batch_mask
  (void)in_sizes; (void)n_in; (void)out_size; (void)ws_size;

  float* out    = (float*)d_out;
  float* out_sm = out;                  // s_m   (512*128)
  float* out_so = out + NA * NF;        // s_out (512*128)
  float* out_vm = out + 2 * NA * NF;    // v_m   (512*3*128)

  char* wsb = (char*)d_ws;
  size_t off = 0;
  auto take = [&](size_t bytes) -> void* {
    void* p = wsb + off;
    off = (off + bytes + 255) & ~(size_t)255;
    return p;
  };
  // f16 transposed weights
  _Float16* phiT = (_Float16*)take(NF * NF * 2);
  _Float16* qT   = (_Float16*)take(NF * NF * 2);
  _Float16* kT   = (_Float16*)take(NF * NF * 2);
  _Float16* vlT  = (_Float16*)take(NF * NF * 2);
  _Float16* r1T  = (_Float16*)take(NF * NF * 2);
  _Float16* r2T  = (_Float16*)take(NF * NF * 2);
  _Float16* w1T  = (_Float16*)take(KP * NF * 2);
  _Float16* w2T  = (_Float16*)take(KP * NF * 2);
  _Float16* ocfT = (_Float16*)take(C2 * NO * 2);
  _Float16* uT   = (_Float16*)take(NF * NO * 2);
  _Float16* oT   = (_Float16*)take(NF * NO * 2);
  // f16 activations (GEMM A operands)
  _Float16* s16    = (_Float16*)take(NA * NF * 2);
  _Float16* v16    = (_Float16*)take(NA * 3 * NF * 2);
  _Float16* spre16 = (_Float16*)take(NA * NF * 2);
  _Float16* sm16   = (_Float16*)take(NA * NF * 2);
  _Float16* h16    = (_Float16*)take(NA * NF * 2);
  // f32 intermediates
  float* xp   = (float*)take(NA * NF * 4);
  float* qv   = (float*)take(NA * NF * 4);
  float* kv   = (float*)take(NA * NF * 4);
  float* vav  = (float*)take(NA * NF * 4);
  float* s_nl = (float*)take(NA * NF * 4);
  float* s1   = (float*)take(NA * NF * 4);
  float* s2   = (float*)take(NA * NF * 4);
  float* vm3  = (float*)take(NA * 3 * NF * 4);
  float* s_n  = (float*)take(NA * NO * 4);
  float* u_o  = (float*)take(NA * 3 * NO * 4);

  auto launchT = [&](const float* W, _Float16* T, int O, int C, int Kp) {
    int n = Kp * O;
    transpose_h_kernel<<<(n + 255) / 256, 256, 0, stream>>>(W, T, O, C, Kp);
  };
  launchT(phi_W, phiT, NF, NF, NF);
  launchT(q_W,   qT,   NF, NF, NF);
  launchT(k_W,   kT,   NF, NF, NF);
  launchT(vl_W,  vlT,  NF, NF, NF);
  launchT(r1_W,  r1T,  NF, NF, NF);
  launchT(r2_W,  r2T,  NF, NF, NF);
  launchT(w1_W,  w1T,  NF, NK, KP);
  launchT(w2_W,  w2T,  NF, NK, KP);
  launchT(ocf_W, ocfT, NO, C2, C2);
  launchT(u_W,   uT,   NO, NF, NF);
  launchT(o_W,   oT,   NO, NF, NF);

  to_half_kernel<<<(NA * NF / 4 + 255) / 256, 256, 0, stream>>>(s, s16, NA * NF / 4);
  to_half_kernel<<<(NA * 3 * NF / 4 + 255) / 256, 256, 0, stream>>>(v, v16, NA * 3 * NF / 4);

  auto launchG = [&](const _Float16* A, const _Float16* Bt, const float* bias,
                     const float* a0, const float* a1, float* C, _Float16* Ch,
                     int M, int N, int act) {
    dim3 g(M / 16, N / 128);
    gemm_h_kernel<<<g, 256, 0, stream>>>(A, Bt, bias, a0, a1, C, Ch, N, act);
  };

  // projections
  launchG(s16, phiT, phi_b, nullptr, nullptr, xp,  nullptr, NA, NF, 1);  // silu
  launchG(s16, qT,   q_b,   nullptr, nullptr, qv,  nullptr, NA, NF, 0);
  launchG(s16, kT,   k_b,   nullptr, nullptr, kv,  nullptr, NA, NF, 0);
  launchG(s16, vlT,  vl_b,  nullptr, nullptr, vav, nullptr, NA, NF, 0);

  attn_kernel<<<NA, 128, 0, stream>>>(qv, kv, vav, bmask, s_nl);

  cfconv_kernel<<<NA, 256, 0, stream>>>(e, vec_norm, mask, xp, w1T, w2T, ocfT,
                                        w1_b, w2_b, ocf_b, s1, s2, vm3);

  spre_kernel<<<(NA * NF) / 256, 256, 0, stream>>>(s, s_nl, s1, spre16);

  launchG(spre16, oT, o_b,    nullptr, nullptr, s_n, nullptr, NA,     NO, 0);
  launchG(v16,    uT, nullptr, nullptr, nullptr, u_o, nullptr, NA * 3, NO, 0);

  combine_kernel<<<NA, 128, 0, stream>>>(s_n, u_o, s2, vm3, v, out_sm, sm16, out_vm);

  // residual MLP: s_out = (silu(s_m@r1T+b1))@r2T + b2 + s_m + o_prev
  launchG(sm16, r1T, r1_b, nullptr, nullptr, nullptr, h16,  NA, NF, 1);
  launchG(h16,  r2T, r2_b, out_sm,  o_prev,  out_so, nullptr, NA, NF, 0);
}
